// TransformerEncoder_33801392619590
// MI455X (gfx1250) — compile-verified, matrix operand order checked by reading, not run on hardware
//
#include <hip/hip_runtime.h>
#include <hip/hip_bf16.h>
#include <math.h>

// ---------------------------------------------------------------------------
// Types for CDNA5 WMMA
// ---------------------------------------------------------------------------
typedef __bf16 bf16_t;
typedef __attribute__((ext_vector_type(16))) __bf16 v16bf;
typedef __attribute__((ext_vector_type(8)))  float  v8f;

static __device__ __forceinline__ bf16_t to_bf16(float f) {
  unsigned u = __builtin_bit_cast(unsigned, f);
  unsigned r = (u + 0x7FFFu + ((u >> 16) & 1u)) >> 16;
  unsigned short s = (unsigned short)r;
  return __builtin_bit_cast(bf16_t, s);
}

static __device__ __forceinline__ v8f wmma_bf16(v16bf a, v16bf b, v8f c) {
  // D = A(16x32 bf16) * B(32x16 bf16) + C(16x16 f32)
  return __builtin_amdgcn_wmma_f32_16x16x32_bf16(
      /*neg_a=*/false, a, /*neg_b=*/false, b,
      /*c_mod=*/(short)0, c, /*reuse_a=*/false, /*reuse_b=*/false);
}

// Load one 16x32 bf16 operand fragment from a K-contiguous matrix
// (rows = the 16 M-or-N indices of the tile, K runs contiguously per row).
// ISA layout (CDNA5 ISA 7.12.2): lane L (0..15) holds row L, K = {0..7,16..23};
// lane L+16 holds row L, K = {8..15,24..31}. Two 16B loads per lane.
static __device__ __forceinline__ v16bf load_frag(const bf16_t* base, int row_stride) {
  int lane = threadIdx.x & 31;
  int L    = lane & 15;
  int half = lane >> 4;
  const bf16_t* p = base + (size_t)L * row_stride + half * 8;
  union { uint4 u[2]; v16bf v; } un;
  un.u[0] = *(const uint4*)(p);        // K = half*8 .. half*8+7
  un.u[1] = *(const uint4*)(p + 16);   // K = 16+half*8 .. 16+half*8+7
  return un.v;
}

// ---------------------------------------------------------------------------
// Constants for this problem
// ---------------------------------------------------------------------------
#define HID   768
#define SEQ   4096
#define BATCH 2
#define NHEAD 8
#define HDIM  96
#define EXPH  3072
#define NTOK  (BATCH * SEQ)   // 8192

// ---------------------------------------------------------------------------
// Weight transpose + convert: wt[n*K + k] = bf16(w[k*N + n])
// ---------------------------------------------------------------------------
__global__ void transpose_w_kernel(const float* __restrict__ w, bf16_t* __restrict__ wt,
                                   int K, int N) {
  int idx = blockIdx.x * blockDim.x + threadIdx.x;
  if (idx >= K * N) return;
  int n = idx / K;
  int k = idx - n * K;
  wt[idx] = to_bf16(w[(size_t)k * N + n]);
}

// ---------------------------------------------------------------------------
// LayerNorm: one 256-thread block per token row (768 elems, 3 per thread)
// ---------------------------------------------------------------------------
__global__ void ln_kernel(const float* __restrict__ x, const float* __restrict__ g,
                          const float* __restrict__ b, bf16_t* __restrict__ out) {
  __shared__ float sred[256];
  int row = blockIdx.x;
  int t   = threadIdx.x;
  const float* xr = x + (size_t)row * HID;
  float v0 = xr[t], v1 = xr[t + 256], v2 = xr[t + 512];
  sred[t] = v0 + v1 + v2;
  __syncthreads();
  for (int o = 128; o > 0; o >>= 1) { if (t < o) sred[t] += sred[t + o]; __syncthreads(); }
  float mu = sred[0] * (1.0f / HID);
  __syncthreads();
  float d0 = v0 - mu, d1 = v1 - mu, d2 = v2 - mu;
  sred[t] = d0 * d0 + d1 * d1 + d2 * d2;
  __syncthreads();
  for (int o = 128; o > 0; o >>= 1) { if (t < o) sred[t] += sred[t + o]; __syncthreads(); }
  float rs = rsqrtf(sred[0] * (1.0f / HID) + 1e-5f);
  bf16_t* orow = out + (size_t)row * HID;
  orow[t]       = to_bf16(d0 * rs * g[t]       + b[t]);
  orow[t + 256] = to_bf16(d1 * rs * g[t + 256] + b[t + 256]);
  orow[t + 512] = to_bf16(d2 * rs * g[t + 512] + b[t + 512]);
}

// ---------------------------------------------------------------------------
// WMMA GEMM: C[M,N] = A[M,K](bf16) @ Bt[N,K](bf16)^T + bias (+residual)(gelu?)
// Block: 256 threads (8 waves, 4x2), tile 128x128, K-step 32.
// ---------------------------------------------------------------------------
#define BM 128
#define BN 128
#define BK 32
#define LDT 40  // padded LDS row stride (bf16 elems), multiple of 8

__global__ __launch_bounds__(256)
void gemm_bf16_kernel(const bf16_t* __restrict__ A, const bf16_t* __restrict__ Bt,
                      const float* __restrict__ bias, const float* __restrict__ residual,
                      float* __restrict__ outF, bf16_t* __restrict__ outB,
                      int M, int N, int K, int act /*0 none, 1 gelu*/) {
  __shared__ __align__(16) bf16_t sA[BM * LDT];
  __shared__ __align__(16) bf16_t sB[BN * LDT];
  int bm = blockIdx.y, bn = blockIdx.x;
  int t = threadIdx.x;
  int wave = t >> 5, lane = t & 31;
  int wm = wave & 3;      // 0..3 -> 32 rows each
  int wn = wave >> 2;     // 0..1 -> 64 cols each
  int L = lane & 15, half = lane >> 4;

  v8f acc[2][4];
  for (int i = 0; i < 2; ++i)
    for (int j = 0; j < 4; ++j) acc[i][j] = v8f{0.f,0.f,0.f,0.f,0.f,0.f,0.f,0.f};

  for (int k0 = 0; k0 < K; k0 += BK) {
    // Stage A and Bt tiles (128 rows x 32 K) into LDS: 512 8-elem chunks each.
    for (int l = t; l < 512; l += 256) {
      int r = l >> 2, kk = (l & 3) << 3;
      *(uint4*)&sA[r * LDT + kk] =
          *(const uint4*)&A[(size_t)(bm * BM + r) * K + k0 + kk];
      if (k0 + BK < K)
        __builtin_prefetch(&A[(size_t)(bm * BM + r) * K + k0 + BK + kk], 0, 1);
    }
    for (int l = t; l < 512; l += 256) {
      int r = l >> 2, kk = (l & 3) << 3;
      *(uint4*)&sB[r * LDT + kk] =
          *(const uint4*)&Bt[(size_t)(bn * BN + r) * K + k0 + kk];
      if (k0 + BK < K)
        __builtin_prefetch(&Bt[(size_t)(bn * BN + r) * K + k0 + BK + kk], 0, 1);
    }
    __syncthreads();

    v16bf af[2], bfr[4];
    for (int i = 0; i < 2; ++i) af[i]  = load_frag(&sA[(wm * 32 + i * 16) * LDT], LDT);
    for (int j = 0; j < 4; ++j) bfr[j] = load_frag(&sB[(wn * 64 + j * 16) * LDT], LDT);
    for (int i = 0; i < 2; ++i)
      for (int j = 0; j < 4; ++j)
        acc[i][j] = wmma_bf16(af[i], bfr[j], acc[i][j]);
    __syncthreads();
  }

  // Epilogue. C layout: VGPR e -> row = e + 8*half, col = L (within 16x16 tile)
  for (int i = 0; i < 2; ++i)
    for (int j = 0; j < 4; ++j) {
      int col = bn * BN + wn * 64 + j * 16 + L;
      float bv = bias[col];
      for (int e = 0; e < 8; ++e) {
        int row = bm * BM + wm * 32 + i * 16 + e + 8 * half;
        float v = acc[i][j][e] + bv;
        if (act == 1) v = 0.5f * v * (1.0f + erff(v * 0.70710678118654752f));
        size_t idx = (size_t)row * N + col;
        if (residual) v += residual[idx];
        if (outF) outF[idx] = v;
        if (outB) outB[idx] = to_bf16(v);
      }
    }
}

// ---------------------------------------------------------------------------
// vt[b][h][d][s] = v[b][s][h*96+d]   (bf16 -> bf16 per-head transpose)
// ---------------------------------------------------------------------------
__global__ void vt_kernel(const bf16_t* __restrict__ v, bf16_t* __restrict__ vt) {
  int idx = blockIdx.x * blockDim.x + threadIdx.x;
  if (idx >= BATCH * NHEAD * HDIM * SEQ) return;
  int s    = idx & (SEQ - 1);
  int rest = idx >> 12;            // (b*8+h)*96 + d
  int d    = rest % HDIM;
  int bh   = rest / HDIM;
  int b    = bh >> 3, h = bh & 7;
  vt[idx] = v[((size_t)(b * SEQ + s)) * HID + h * HDIM + d];
}

// ---------------------------------------------------------------------------
// Pass A: denom[b][q][k] = sum_h exp(q_h . k_h)
// Grid (ktile=128, qtile=128, b=2); 256 threads, wave = head.
// Each wave: 32x32 energy tile, K=96 in 3 WMMA steps, exp -> LDS slab -> sum.
// ---------------------------------------------------------------------------
__global__ __launch_bounds__(256)
void denom_kernel(const bf16_t* __restrict__ q, const bf16_t* __restrict__ k,
                  float* __restrict__ denom) {
  __shared__ float slab[NHEAD][1024];   // 32KB
  int ktile = blockIdx.x, qtile = blockIdx.y, bidx = blockIdx.z;
  int wave = threadIdx.x >> 5, lane = threadIdx.x & 31;
  int h = wave;
  int L = lane & 15, half = lane >> 4;
  int q0 = qtile * 32, k0 = ktile * 32;

  const bf16_t* qb = q + (size_t)bidx * SEQ * HID + h * HDIM;
  const bf16_t* kb = k + (size_t)bidx * SEQ * HID + h * HDIM;

  v8f acc[2][2];
  for (int i = 0; i < 2; ++i)
    for (int j = 0; j < 2; ++j) acc[i][j] = v8f{0.f,0.f,0.f,0.f,0.f,0.f,0.f,0.f};

  for (int c = 0; c < 3; ++c) {
    v16bf aq[2], bk[2];
    for (int i = 0; i < 2; ++i)
      aq[i] = load_frag(qb + (size_t)(q0 + i * 16) * HID + c * 32, HID);
    for (int j = 0; j < 2; ++j)
      bk[j] = load_frag(kb + (size_t)(k0 + j * 16) * HID + c * 32, HID);
    for (int i = 0; i < 2; ++i)
      for (int j = 0; j < 2; ++j)
        acc[i][j] = wmma_bf16(aq[i], bk[j], acc[i][j]);
  }

  for (int i = 0; i < 2; ++i)
    for (int j = 0; j < 2; ++j)
      for (int e = 0; e < 8; ++e) {
        int r = i * 16 + e + 8 * half;   // q row within 32
        int c = j * 16 + L;              // k col within 32
        slab[h][r * 32 + c] = __expf(acc[i][j][e]);
      }
  __syncthreads();

  float* drow = denom + (size_t)bidx * SEQ * SEQ;
  for (int idx = threadIdx.x; idx < 1024; idx += 256) {
    float s = 0.f;
    for (int hh = 0; hh < NHEAD; ++hh) s += slab[hh][idx];
    int r = idx >> 5, c = idx & 31;
    drow[(size_t)(q0 + r) * SEQ + (k0 + c)] = s;
  }
}

// ---------------------------------------------------------------------------
// Pass B: attn_out[b][q][h*96+d] = sum_k (exp(e)/denom/sqrt(768)) * v
// Grid (qblock=64, h=8, b=2); 128 threads = 4 waves; each wave owns 16 q rows
// and streams all 128 k-tiles, transposing att through a wave-private LDS slab.
// ---------------------------------------------------------------------------
__global__ __launch_bounds__(128)
void attn_kernel(const bf16_t* __restrict__ q, const bf16_t* __restrict__ k,
                 const bf16_t* __restrict__ vt, const float* __restrict__ denom,
                 bf16_t* __restrict__ attn_out) {
  __shared__ __align__(16) bf16_t satt[4][16 * LDT];
  int qblock = blockIdx.x, h = blockIdx.y, bidx = blockIdx.z;
  int wave = threadIdx.x >> 5, lane = threadIdx.x & 31;
  int L = lane & 15, half = lane >> 4;
  int q0 = qblock * 64 + wave * 16;

  const bf16_t* qb  = q  + ((size_t)bidx * SEQ + q0) * HID + h * HDIM;
  const bf16_t* kb  = k  + (size_t)bidx * SEQ * HID + h * HDIM;
  const bf16_t* vtb = vt + (size_t)(bidx * NHEAD + h) * HDIM * SEQ;
  const float*  db  = denom + (size_t)bidx * SEQ * SEQ + (size_t)q0 * SEQ;
  bf16_t* sw = satt[wave];

  v16bf aq[3];
  for (int c = 0; c < 3; ++c) aq[c] = load_frag(qb + c * 32, HID);

  v8f out[6];
  for (int n = 0; n < 6; ++n) out[n] = v8f{0.f,0.f,0.f,0.f,0.f,0.f,0.f,0.f};

  const float scale = 0.03608439182435161f;  // 1/sqrt(768)

  for (int kt = 0; kt < SEQ / 32; ++kt) {
    int kpos0 = kt * 32;
    v8f e[2];
    e[0] = v8f{0.f,0.f,0.f,0.f,0.f,0.f,0.f,0.f};
    e[1] = v8f{0.f,0.f,0.f,0.f,0.f,0.f,0.f,0.f};
    for (int c = 0; c < 3; ++c) {
      v16bf bk0 = load_frag(kb + (size_t)kpos0 * HID + c * 32, HID);
      v16bf bk1 = load_frag(kb + (size_t)(kpos0 + 16) * HID + c * 32, HID);
      e[0] = wmma_bf16(aq[c], bk0, e[0]);
      e[1] = wmma_bf16(aq[c], bk1, e[1]);
    }
    // att = exp(e) / (denom * sqrt(768)); write to LDS in A-fragment layout
    for (int j = 0; j < 2; ++j)
      for (int eI = 0; eI < 8; ++eI) {
        int r = eI + 8 * half;        // q row within 16
        int c = j * 16 + L;           // k col within 32
        float dv = db[(size_t)r * SEQ + kpos0 + c];
        float a  = __expf(e[j][eI]) * scale / dv;
        sw[r * LDT + c] = to_bf16(a);
      }
    // same-wave DS ops are in-order; the wait + memory clobber stops the
    // compiler from hoisting the fragment loads above the stores.
    asm volatile("s_wait_dscnt 0" ::: "memory");
    v16bf afrag = load_frag(sw, LDT);
    for (int n = 0; n < 6; ++n) {
      v16bf vfrag = load_frag(vtb + (size_t)(n * 16) * SEQ + kpos0, SEQ);
      out[n] = wmma_bf16(afrag, vfrag, out[n]);
    }
  }

  bf16_t* ob = attn_out + ((size_t)bidx * SEQ + q0) * HID + h * HDIM;
  for (int n = 0; n < 6; ++n)
    for (int eI = 0; eI < 8; ++eI) {
      int r = eI + 8 * half;
      ob[(size_t)r * HID + n * 16 + L] = to_bf16(out[n][eI]);
    }
}

// ---------------------------------------------------------------------------
// Host-side launch
// ---------------------------------------------------------------------------
extern "C" void kernel_launch(void* const* d_in, const int* in_sizes, int n_in,
                              void* d_out, int out_size, void* d_ws, size_t ws_size,
                              hipStream_t stream) {
  const float* x     = (const float*)d_in[0];
  const float* ln1_g = (const float*)d_in[1];
  const float* ln1_b = (const float*)d_in[2];
  const float* wq    = (const float*)d_in[3];
  const float* bq    = (const float*)d_in[4];
  const float* wk    = (const float*)d_in[5];
  const float* bk    = (const float*)d_in[6];
  const float* wv    = (const float*)d_in[7];
  const float* bv    = (const float*)d_in[8];
  const float* wp    = (const float*)d_in[9];
  const float* bp    = (const float*)d_in[10];
  const float* ln2_g = (const float*)d_in[11];
  const float* ln2_b = (const float*)d_in[12];
  const float* w1    = (const float*)d_in[13];
  const float* b1    = (const float*)d_in[14];
  const float* w2    = (const float*)d_in[15];
  const float* b2    = (const float*)d_in[16];
  float* out = (float*)d_out;

  // Workspace carve-up (aligned to 256B).
  char* ws = (char*)d_ws;
  size_t off = 0;
  auto alloc = [&](size_t bytes) -> void* {
    void* p = ws + off;
    off += (bytes + 255) & ~(size_t)255;
    return p;
  };
  const size_t tokH  = (size_t)NTOK * HID;
  bf16_t* xn1  = (bf16_t*)alloc(tokH * 2);
  bf16_t* xn2  = (bf16_t*)alloc(tokH * 2);
  bf16_t* qb   = (bf16_t*)alloc(tokH * 2);
  bf16_t* kbuf = (bf16_t*)alloc(tokH * 2);
  bf16_t* vbuf = (bf16_t*)alloc(tokH * 2);
  bf16_t* vtb  = (bf16_t*)alloc(tokH * 2);
  bf16_t* attb = (bf16_t*)alloc(tokH * 2);
  bf16_t* hmid = (bf16_t*)alloc((size_t)NTOK * EXPH * 2);
  float*  x1   = (float*)alloc(tokH * 4);
  float*  den  = (float*)alloc((size_t)BATCH * SEQ * SEQ * 4);
  bf16_t* wqT  = (bf16_t*)alloc((size_t)HID * HID * 2);
  bf16_t* wkT  = (bf16_t*)alloc((size_t)HID * HID * 2);
  bf16_t* wvT  = (bf16_t*)alloc((size_t)HID * HID * 2);
  bf16_t* wpT  = (bf16_t*)alloc((size_t)HID * HID * 2);
  bf16_t* w1T  = (bf16_t*)alloc((size_t)HID * EXPH * 2);
  bf16_t* w2T  = (bf16_t*)alloc((size_t)HID * EXPH * 2);

  // 1. Weight transposes (f32 -> bf16, [K,N] -> [N,K]).
  {
    int n1 = HID * HID;
    transpose_w_kernel<<<(n1 + 255) / 256, 256, 0, stream>>>(wq, wqT, HID, HID);
    transpose_w_kernel<<<(n1 + 255) / 256, 256, 0, stream>>>(wk, wkT, HID, HID);
    transpose_w_kernel<<<(n1 + 255) / 256, 256, 0, stream>>>(wv, wvT, HID, HID);
    transpose_w_kernel<<<(n1 + 255) / 256, 256, 0, stream>>>(wp, wpT, HID, HID);
    int n2 = HID * EXPH;
    transpose_w_kernel<<<(n2 + 255) / 256, 256, 0, stream>>>(w1, w1T, HID, EXPH);
    transpose_w_kernel<<<(n2 + 255) / 256, 256, 0, stream>>>(w2, w2T, EXPH, HID);
  }

  // 2. LN1
  ln_kernel<<<NTOK, 256, 0, stream>>>(x, ln1_g, ln1_b, xn1);

  // 3. Q/K/V projections (bf16 outputs)
  dim3 g768(HID / BN, NTOK / BM);
  gemm_bf16_kernel<<<g768, 256, 0, stream>>>(xn1, wqT, bq, nullptr, nullptr, qb,
                                             NTOK, HID, HID, 0);
  gemm_bf16_kernel<<<g768, 256, 0, stream>>>(xn1, wkT, bk, nullptr, nullptr, kbuf,
                                             NTOK, HID, HID, 0);
  gemm_bf16_kernel<<<g768, 256, 0, stream>>>(xn1, wvT, bv, nullptr, nullptr, vbuf,
                                             NTOK, HID, HID, 0);

  // 4. Per-head transpose of V
  {
    int n = BATCH * NHEAD * HDIM * SEQ;
    vt_kernel<<<(n + 255) / 256, 256, 0, stream>>>(vbuf, vtb);
  }

  // 5. Head-axis softmax denominators
  denom_kernel<<<dim3(SEQ / 32, SEQ / 32, BATCH), 256, 0, stream>>>(qb, kbuf, den);

  // 6. Attention output (streaming att@v)
  attn_kernel<<<dim3(SEQ / 64, NHEAD, BATCH), 128, 0, stream>>>(qb, kbuf, vtb, den, attb);

  // 7. Output projection + residual -> x1 (f32)
  gemm_bf16_kernel<<<g768, 256, 0, stream>>>(attb, wpT, bp, x, x1, nullptr,
                                             NTOK, HID, HID, 0);

  // 8. LN2
  ln_kernel<<<NTOK, 256, 0, stream>>>(x1, ln2_g, ln2_b, xn2);

  // 9. MLP up + exact GELU -> hmid (bf16)
  dim3 g3072(EXPH / BN, NTOK / BM);
  gemm_bf16_kernel<<<g3072, 256, 0, stream>>>(xn2, w1T, b1, nullptr, nullptr, hmid,
                                              NTOK, EXPH, HID, 1);

  // 10. MLP down + residual -> d_out (f32)
  gemm_bf16_kernel<<<g768, 256, 0, stream>>>(hmid, w2T, b2, x1, out, nullptr,
                                             NTOK, HID, EXPH, 0);
}
